// Visual_model_6786048328135
// MI455X (gfx1250) — compile-verified, matrix operand order checked by reading, not run on hardware
//
#include <hip/hip_runtime.h>
#include <hip/hip_bf16.h>

#define B_  16
#define T_  1024
#define D_  2048
#define H0_ 1024
#define H1_ 256
#define LR_ 100
#define ROWS_ 32            // M-tile per workgroup
#define PASSCOLS_ 512       // GEMM1 N columns staged per pass
#define KT_ 32              // K per staged tile
#define BSTRIDE_ 80         // bytes per staged column (64B data + 16B TDM pad)

typedef __attribute__((ext_vector_type(16))) __bf16 v16bf;
typedef __attribute__((ext_vector_type(8)))  __bf16 v8bf;
typedef __attribute__((ext_vector_type(8)))  float  v8f;
typedef unsigned int u32x4 __attribute__((ext_vector_type(4)));
typedef int          i32x4 __attribute__((ext_vector_type(4)));
typedef int          i32x8 __attribute__((ext_vector_type(8)));

__device__ __forceinline__ unsigned short f32_to_bf16_rne(float f) {
    unsigned int u = __float_as_uint(f);
    u += 0x7FFFu + ((u >> 16) & 1u);
    return (unsigned short)(u >> 16);
}
__device__ __forceinline__ float bf16_to_f32(unsigned short h) {
    return __uint_as_float(((unsigned int)h) << 16);
}
__device__ __forceinline__ float sigmoidf_(float x) {
    return 1.0f / (1.0f + __expf(-x));
}

// TDM: load a (KT_ x PASSCOLS_) bf16 tile of W0t (column-major, k contiguous,
// column stride D_) into LDS with 16B padding per 64B line -> 80B col stride.
// D# per CDNA5 ISA 8.3/8.4. Uniform args; issued by one wave, TENSORcnt-tracked.
__device__ __forceinline__ void tdm_load_tile(const unsigned short* gptr, void* lptr) {
    unsigned long long ga = (unsigned long long)(uintptr_t)gptr;
    unsigned int lds = (unsigned int)(uintptr_t)lptr;   // flat LDS addr low 32 = LDS byte offset
    u32x4 g0;
    g0[0] = 1u;                                              // count=1 (valid user D#)
    g0[1] = lds;                                             // lds_addr
    g0[2] = (unsigned int)(ga & 0xFFFFFFFFu);                // global_addr[31:0]
    g0[3] = (unsigned int)((ga >> 32) & 0x1FFFFFFu)          // global_addr[56:32]
          | (2u << 30);                                      // type = 2 ("image")
    i32x8 g1;
    g1[0] = (int)((1u << 16)      // data_size = 2 bytes
          |       (1u << 20)      // pad_enable
          |       (3u << 22)      // pad_interval: 16 dwords (64B)
          |       (3u << 25));    // pad_amount: 4 dwords (16B)
    g1[1] = (int)((unsigned)D_ << 16);     // tensor_dim0 lo16 = 2048
    g1[2] = (int)((unsigned)H0_ << 16);    // tensor_dim0 hi = 0 | tensor_dim1 lo16 = 1024
    g1[3] = (int)((unsigned)KT_ << 16);    // tensor_dim1 hi = 0 | tile_dim0 = 32
    g1[4] = (int)PASSCOLS_;                // tile_dim1 = 512, tile_dim2 = 0
    g1[5] = (int)D_;                       // tensor_dim0_stride lo32 = 2048
    g1[6] = (int)((unsigned)D_ << 16);     // stride0 hi16 = 0 | tensor_dim1_stride lo16 = 2048
    g1[7] = 0;                             // tensor_dim1_stride hi
    i32x4 z4 = {0, 0, 0, 0};
#if defined(__clang_major__) && __clang_major__ >= 23
    i32x8 z8 = {0, 0, 0, 0, 0, 0, 0, 0};
    __builtin_amdgcn_tensor_load_to_lds(g0, g1, z4, z4, z8, 0);
#else
    __builtin_amdgcn_tensor_load_to_lds(g0, g1, z4, z4, 0);
#endif
}

// One K-tile worth of GEMM1 WMMAs from a fixed staging buffer.
__device__ __forceinline__ void gemm1_step(
    v8f acc0[4], v8f acc1[4],
    const unsigned short* arow0, const unsigned short* arow1,
    const char* buf, int ko, int colbase, int half)
{
    v8bf a0lo = *(const v8bf*)(arow0 + ko);
    v8bf a0hi = *(const v8bf*)(arow0 + ko + 16);
    v16bf a0 = __builtin_shufflevector(a0lo, a0hi,
                 0,1,2,3,4,5,6,7,8,9,10,11,12,13,14,15);
    v8bf a1lo = *(const v8bf*)(arow1 + ko);
    v8bf a1hi = *(const v8bf*)(arow1 + ko + 16);
    v16bf a1 = __builtin_shufflevector(a1lo, a1hi,
                 0,1,2,3,4,5,6,7,8,9,10,11,12,13,14,15);
    #pragma unroll
    for (int j = 0; j < 4; ++j) {
        const char* bp = buf + (size_t)(colbase + j * 16) * BSTRIDE_ + half * 16;
        v8bf blo = *(const v8bf*)(bp);
        v8bf bhi = *(const v8bf*)(bp + 32);
        v16bf bm = __builtin_shufflevector(blo, bhi,
                    0,1,2,3,4,5,6,7,8,9,10,11,12,13,14,15);
        acc0[j] = __builtin_amdgcn_wmma_f32_16x16x32_bf16(
                    false, a0, false, bm, (short)0, acc0[j], false, false);
        acc1[j] = __builtin_amdgcn_wmma_f32_16x16x32_bf16(
                    false, a1, false, bm, (short)0, acc1[j], false, false);
    }
}

// ---------------------------------------------------------------------------
// Kernel 0: W0/W1 -> bf16 transposed in workspace; zero accumulators.
// ---------------------------------------------------------------------------
__global__ __launch_bounds__(256) void prep_kernel(
    const float* __restrict__ W0, const float* __restrict__ W1,
    unsigned short* __restrict__ W0t, unsigned short* __restrict__ W1t,
    float* __restrict__ accW, float* __restrict__ accP,
    float* __restrict__ accN, float* __restrict__ cnts)
{
    const int i0 = blockIdx.x * blockDim.x + threadIdx.x;
    const int stride = gridDim.x * blockDim.x;
    for (int i = i0; i < D_ * H0_; i += stride) {
        const int k = i >> 10;
        const int c = i & (H0_ - 1);
        W0t[(size_t)c * D_ + k] = f32_to_bf16_rne(W0[i]);
    }
    for (int i = i0; i < H0_ * H1_; i += stride) {
        const int k = i >> 8;
        const int c = i & (H1_ - 1);
        W1t[(size_t)c * H0_ + k] = f32_to_bf16_rne(W1[i]);
    }
    for (int i = i0; i < B_ * D_; i += stride) {
        accW[i] = 0.f; accP[i] = 0.f; accN[i] = 0.f;
    }
    for (int i = i0; i < 3 * B_; i += stride) cnts[i] = 0.f;
}

// ---------------------------------------------------------------------------
// Kernel 1: fused MLP + attention + pooled sums.
// GEMM1 streams W0 K-tiles through LDS via the Tensor Data Mover, double-
// buffered (TENSORcnt); kt-loop unrolled by 2 with fixed buffer pointers so
// the WMMA accumulators stay stationary (no rotating-register mov blocks).
// ---------------------------------------------------------------------------
__global__ __launch_bounds__(256) void fused_mlp_att(
    const float* __restrict__ feats, const int* __restrict__ props,
    const float* __restrict__ text,  const float* __restrict__ b0,
    const float* __restrict__ b1,    const float* __restrict__ W2,
    const float* __restrict__ b2,
    const unsigned short* __restrict__ W0t,
    const unsigned short* __restrict__ W1t,
    float* __restrict__ att_out, float* __restrict__ accW,
    float* __restrict__ accP,    float* __restrict__ accN,
    float* __restrict__ cnts)
{
    extern __shared__ char smem[];
    const int SA = D_ + 8;    // bf16 elems per A-tile row
    const int SH = H0_ + 8;   // bf16 elems per h row
    const int S2 = H1_ + 4;   // f32 elems per h2 row

    // LDS layout (bytes): lA 131584 | lH 66048 | misc 512 | lB0 40960 | lB1 40960
    unsigned short* lA  = (unsigned short*)smem;
    unsigned short* lH  = (unsigned short*)(smem + ROWS_ * SA * 2);
    float*          lH2 = (float*)(smem + ROWS_ * SA * 2);              // overlaps lH
    unsigned int*   flags = (unsigned int*)(smem + ROWS_ * SA * 2 + ROWS_ * SH * 2);
    int*            propL = (int*)(flags + ROWS_);
    float*          attw  = (float*)(propL + ROWS_);
    char*           lB0 = smem + ROWS_ * SA * 2 + ROWS_ * SH * 2 + 512;
    char*           lB1 = lB0 + PASSCOLS_ * BSTRIDE_;

    const int tid  = threadIdx.x;
    const int tile = blockIdx.x;
    const int b    = tile >> 5;           // 32 tiles per batch row (T/32)
    const int t0   = (tile & 31) << 5;

    if (tid < ROWS_) {
        flags[tid] = 0u;
        propL[tid] = props[b * T_ + t0 + tid];
    }
    __syncthreads();

    // -------- Phase 1: feature tile -> bf16 LDS, valid-row detection --------
    {
        const int r   = tid >> 3;         // 0..31
        const int seg = tid & 7;          // 8 threads per row, 256 elems each
        const float4* src = (const float4*)(feats + (size_t)(b * T_ + t0 + r) * D_ + seg * 256);
        unsigned short* dst = lA + r * SA + seg * 256;
        int anynz = 0;
        #pragma unroll 4
        for (int i = 0; i < 64; ++i) {
            float4 v = src[i];
            anynz |= (v.x != 0.f) | (v.y != 0.f) | (v.z != 0.f) | (v.w != 0.f);
            unsigned int p0 = (unsigned int)f32_to_bf16_rne(v.x) |
                              ((unsigned int)f32_to_bf16_rne(v.y) << 16);
            unsigned int p1 = (unsigned int)f32_to_bf16_rne(v.z) |
                              ((unsigned int)f32_to_bf16_rne(v.w) << 16);
            *(uint2*)(dst + i * 4) = make_uint2(p0, p1);
        }
        if (anynz) atomicOr(&flags[r], 1u);
    }
    __syncthreads();

    const int wave = tid >> 5;
    const int lane = tid & 31;
    const int half = lane >> 4;
    const int l16  = lane & 15;
    const v8f zero8 = {0.f, 0.f, 0.f, 0.f, 0.f, 0.f, 0.f, 0.f};
    const int NKT = D_ / KT_;             // 64 staged K-tiles per pass
    const int colbase = wave * 64 + l16;  // staged-buffer column base per lane

    // -------- Phase 2: GEMM1 (32 x 2048) @ (2048 x 1024), TDM-staged B ------
    for (int pass = 0; pass < 2; ++pass) {
        v8f acc0[4], acc1[4];
        #pragma unroll
        for (int j = 0; j < 4; ++j) { acc0[j] = zero8; acc1[j] = zero8; }

        const unsigned short* arow0 = lA + l16 * SA;
        const unsigned short* arow1 = lA + (l16 + 16) * SA;
        const unsigned short* gW = W0t + (size_t)pass * PASSCOLS_ * D_;

        if (wave == 0) {
            tdm_load_tile(gW, lB0);                      // prologue: tile 0
            __builtin_amdgcn_s_wait_tensorcnt(0);
        }
        __syncthreads();

        for (int kt = 0; kt < NKT; kt += 2) {
            // even step: compute from lB0, prefetch kt+1 into lB1
            if (wave == 0)
                tdm_load_tile(gW + (kt + 1) * KT_, lB1);
            gemm1_step(acc0, acc1, arow0, arow1, lB0,
                       kt * KT_ + half * 8, colbase, half);
            if (wave == 0)
                __builtin_amdgcn_s_wait_tensorcnt(0);
            __syncthreads();

            // odd step: compute from lB1, prefetch kt+2 into lB0
            if (wave == 0 && kt + 2 < NKT)
                tdm_load_tile(gW + (kt + 2) * KT_, lB0);
            gemm1_step(acc0, acc1, arow0, arow1, lB1,
                       (kt + 1) * KT_ + half * 8, colbase, half);
            if (wave == 0 && kt + 2 < NKT)
                __builtin_amdgcn_s_wait_tensorcnt(0);
            __syncthreads();
        }

        // bias + relu -> h (bf16 in LDS) for this pass's 512 columns
        #pragma unroll
        for (int j = 0; j < 4; ++j) {
            const int ncol = pass * PASSCOLS_ + wave * 64 + j * 16 + l16;
            const float bias = b0[ncol];
            #pragma unroll
            for (int v = 0; v < 8; ++v) {
                float h0 = acc0[j][v] + bias;
                h0 = h0 > 0.f ? h0 : 0.f;
                lH[(v + half * 8) * SH + ncol] = f32_to_bf16_rne(h0);
                float h1 = acc1[j][v] + bias;
                h1 = h1 > 0.f ? h1 : 0.f;
                lH[(16 + v + half * 8) * SH + ncol] = f32_to_bf16_rne(h1);
            }
        }
    }
    __syncthreads();

    // -------- Phase 3: GEMM2  (32 x 1024) @ (1024 x 256), bf16 WMMA --------
    v8f acc2a[2], acc2b[2];
    acc2a[0] = zero8; acc2a[1] = zero8;
    acc2b[0] = zero8; acc2b[1] = zero8;
    {
        const unsigned short* arow0 = lH + l16 * SH;
        const unsigned short* arow1 = lH + (l16 + 16) * SH;
        const unsigned short* bcol0 = W1t + (size_t)(wave * 32 + l16) * H0_;
        for (int kk = 0; kk < H0_; kk += 32) {
            const int ko = kk + half * 8;
            v8bf a0lo = *(const v8bf*)(arow0 + ko);
            v8bf a0hi = *(const v8bf*)(arow0 + ko + 16);
            v16bf a0 = __builtin_shufflevector(a0lo, a0hi,
                         0,1,2,3,4,5,6,7,8,9,10,11,12,13,14,15);
            v8bf a1lo = *(const v8bf*)(arow1 + ko);
            v8bf a1hi = *(const v8bf*)(arow1 + ko + 16);
            v16bf a1 = __builtin_shufflevector(a1lo, a1hi,
                         0,1,2,3,4,5,6,7,8,9,10,11,12,13,14,15);
            __builtin_prefetch(bcol0 + ko + 64, 0, 1);   // global_prefetch_b8
            #pragma unroll
            for (int j = 0; j < 2; ++j) {
                const unsigned short* bp = bcol0 + (size_t)j * 16 * H0_ + ko;
                v8bf blo = *(const v8bf*)(bp);
                v8bf bhi = *(const v8bf*)(bp + 16);
                v16bf bm = __builtin_shufflevector(blo, bhi,
                            0,1,2,3,4,5,6,7,8,9,10,11,12,13,14,15);
                acc2a[j] = __builtin_amdgcn_wmma_f32_16x16x32_bf16(
                            false, a0, false, bm, (short)0, acc2a[j], false, false);
                acc2b[j] = __builtin_amdgcn_wmma_f32_16x16x32_bf16(
                            false, a1, false, bm, (short)0, acc2b[j], false, false);
            }
        }
    }
    __syncthreads();   // all waves done reading lH; safe to overwrite with lH2
    // bias, language gate, relu -> h2 (f32 in LDS), rows 0..31
    #pragma unroll
    for (int j = 0; j < 2; ++j) {
        const int col  = wave * 32 + j * 16 + l16;
        const float bias = b1[col];
        const float lg   = sigmoidf_(text[b * H1_ + col]);
        #pragma unroll
        for (int v = 0; v < 8; ++v) {
            float h0 = (acc2a[j][v] + bias) * lg;
            h0 = h0 > 0.f ? h0 : 0.f;
            lH2[(v + half * 8) * S2 + col] = h0;
            float h1 = (acc2b[j][v] + bias) * lg;
            h1 = h1 > 0.f ? h1 : 0.f;
            lH2[(16 + v + half * 8) * S2 + col] = h1;
        }
    }
    __syncthreads();

    // -------- Phase 4: attention scalar per row + counts --------
    if (tid < ROWS_) {
        const int r = tid;
        float s = 0.f;
        for (int c = 0; c < H1_; ++c) s += lH2[r * S2 + c] * W2[c];
        float aw = sigmoidf_(s + b2[0]);
        aw = flags[r] ? aw : 0.f;
        attw[r] = aw;
        att_out[b * T_ + t0 + r] = aw;
    }
    if (tid == 0) {
        float nv = 0.f, np = 0.f, nn = 0.f;
        for (int r = 0; r < ROWS_; ++r) {
            if (flags[r]) {
                nv += 1.f;
                if (propL[r] == 1) np += 1.f; else nn += 1.f;
            }
        }
        atomicAdd(&cnts[b],          nv);
        atomicAdd(&cnts[B_ + b],     np);
        atomicAdd(&cnts[2 * B_ + b], nn);
    }
    __syncthreads();

    // -------- Phase 5: pooled feature sums (att-weighted / pos / neg) --------
    {
        const int d0 = tid * 8;
        float sw[8], sp[8], sn[8];
        #pragma unroll
        for (int q = 0; q < 8; ++q) { sw[q] = 0.f; sp[q] = 0.f; sn[q] = 0.f; }
        for (int r = 0; r < ROWS_; ++r) {
            const float aw = attw[r];
            const float pm = (flags[r] && propL[r] == 1) ? 1.f : 0.f;
            const float nm = (flags[r] && propL[r] == 0) ? 1.f : 0.f;
            const unsigned short* xrow = lA + r * SA + d0;
            #pragma unroll
            for (int q = 0; q < 8; ++q) {
                const float x = bf16_to_f32(xrow[q]);
                sw[q] += aw * x; sp[q] += pm * x; sn[q] += nm * x;
            }
        }
        float* aW = accW + (size_t)b * D_ + d0;
        float* aP = accP + (size_t)b * D_ + d0;
        float* aN = accN + (size_t)b * D_ + d0;
        #pragma unroll
        for (int q = 0; q < 8; ++q) {
            atomicAdd(aW + q, sw[q]);   // global_atomic_add_f32
            atomicAdd(aP + q, sp[q]);
            atomicAdd(aN + q, sn[q]);
        }
    }
}

// ---------------------------------------------------------------------------
// Kernel 2: finalize the three pooled projections through Wlr.
// ---------------------------------------------------------------------------
__global__ __launch_bounds__(128) void head_kernel(
    const float* __restrict__ Wlr, const float* __restrict__ blr,
    const float* __restrict__ accW, const float* __restrict__ accP,
    const float* __restrict__ accN, const float* __restrict__ cnts,
    float* __restrict__ out)
{
    const int b = blockIdx.x;
    const int o = threadIdx.x;
    if (o >= LR_) return;
    const float isl = 1.f / cnts[b];
    const float icp = 1.f / cnts[B_ + b];
    const float icn = 1.f / cnts[2 * B_ + b];
    const float* aW = accW + (size_t)b * D_;
    const float* aP = accP + (size_t)b * D_;
    const float* aN = accN + (size_t)b * D_;
    float so = 0.f, sp = 0.f, sn = 0.f;
    for (int d = 0; d < D_; ++d) {
        const float wl = Wlr[(size_t)d * LR_ + o];
        so += aW[d] * wl; sp += aP[d] * wl; sn += aN[d] * wl;
    }
    float* base = out + B_ * T_;
    base[b * LR_ + o]                = so * isl + blr[o];
    base[B_ * LR_ + b * LR_ + o]     = sp * icp + blr[o];
    base[2 * B_ * LR_ + b * LR_ + o] = sn * icn + blr[o];
}

extern "C" void kernel_launch(void* const* d_in, const int* in_sizes, int n_in,
                              void* d_out, int out_size, void* d_ws, size_t ws_size,
                              hipStream_t stream) {
    const float* feats = (const float*)d_in[0];
    const int*   props = (const int*)d_in[1];
    const float* text  = (const float*)d_in[2];
    const float* W0    = (const float*)d_in[3];
    const float* b0    = (const float*)d_in[4];
    const float* W1    = (const float*)d_in[5];
    const float* b1    = (const float*)d_in[6];
    const float* W2    = (const float*)d_in[7];
    const float* b2    = (const float*)d_in[8];
    const float* Wlr   = (const float*)d_in[9];
    const float* blr   = (const float*)d_in[10];
    float* out = (float*)d_out;

    char* ws = (char*)d_ws;
    unsigned short* W0t = (unsigned short*)ws;
    unsigned short* W1t = (unsigned short*)(ws + (size_t)D_ * H0_ * 2);
    float* accW = (float*)(ws + (size_t)D_ * H0_ * 2 + (size_t)H0_ * H1_ * 2);
    float* accP = accW + B_ * D_;
    float* accN = accP + B_ * D_;
    float* cnts = accN + B_ * D_;

    prep_kernel<<<2048, 256, 0, stream>>>(W0, W1, W0t, W1t, accW, accP, accN, cnts);

    const size_t smem = (size_t)ROWS_ * (D_ + 8) * 2    // A tile       131584 B
                      + (size_t)ROWS_ * (H0_ + 8) * 2   // h / h2 ovl    66048 B
                      + 512                             // flags/props/attw
                      + 2 * (size_t)PASSCOLS_ * BSTRIDE_; // TDM B bufs  81920 B
    fused_mlp_att<<<(B_ * T_) / ROWS_, 256, smem, stream>>>(
        feats, props, text, b0, b1, W2, b2, W0t, W1t,
        out, accW, accP, accN, cnts);

    head_kernel<<<B_, 128, 0, stream>>>(Wlr, blr, accW, accP, accN, cnts, out);
}
